// Maj3_13073880449248
// MI455X (gfx1250) — 8-bit matrix operand order repaired and checked by reading, compiled, not run on hardware
//
#include <hip/hip_runtime.h>
#include <hip/hip_bf16.h>
#include <stdint.h>


typedef int v8i __attribute__((ext_vector_type(8)));

#define N_IMG   64
#define C_IN    256
#define C_OUT   256
#define H_DIM   32
#define W_DIM   32

#define W1_BYTES (576u * 1024u)
#define W2_BYTES (192u * 1024u)
#define XB_BYTES (64u * 32u * 32u * 256u)

__global__ __launch_bounds__(256)
void maj3_pack_weights(const float* __restrict__ w,
                       char* __restrict__ wp1, char* __restrict__ wp2) {
    int f    = blockIdx.x;
    int lane = threadIdx.x >> 3;
    int v    = threadIdx.x & 7;
    int m    = lane & 15;
    int kBase = ((v >> 2) << 5) + ((lane >> 4) << 4) + ((v & 3) << 2);

    if (f < 576) {
        int q   = f & 3;
        int t   = (f >> 2) % 9;
        int o16 = f / 36;
        int oc  = o16 * 16 + m;
        int r = t / 3, k = t % 3;
        uint32_t d = 0;
        for (int b = 0; b < 4; ++b) {
            int c = q * 64 + kBase + b;
            float val = w[(((size_t)oc * 3 + r) * 3 + k) * C_IN + c];
            int s = (val > 0.f) - (val < 0.f);
            d |= ((uint32_t)(uint8_t)(int8_t)s) << (8 * b);
        }
        ((uint32_t*)wp1)[(size_t)f * 256 + lane * 8 + v] = d;
    } else {
        int f2  = f - 576;
        int q   = f2 & 3;
        int r   = (f2 >> 2) % 3;
        int o16 = f2 / 12;
        int oc  = o16 * 16 + m;
        uint32_t d = 0;
        for (int b = 0; b < 4; ++b) {
            int c = q * 64 + kBase + b;
            int s = 1;
            for (int k = 0; k < 3; ++k) {
                float val = w[(((size_t)oc * 3 + r) * 3 + k) * C_IN + c];
                s *= (val > 0.f) - (val < 0.f);
            }
            s = -s;
            d |= ((uint32_t)(uint8_t)(int8_t)s) << (8 * b);
        }
        ((uint32_t*)wp2)[(size_t)f2 * 256 + lane * 8 + v] = d;
    }
}

__global__ __launch_bounds__(256)
void maj3_pack_acts(const float* __restrict__ x,
                    char* __restrict__ xbp, char* __restrict__ xpp) {
    __shared__ char ssign[32 * 256];
    __shared__ char sprod[32 * 256];
    int blk = blockIdx.x;
    int n = blk >> 5, i = blk & 31;
    int t = threadIdx.x;
    int j  = t & 31;
    int cb = t >> 5;
    for (int k = 0; k < 32; ++k) {
        int c = cb + k * 8;
        float val = x[(((size_t)(n * C_IN + c) * H_DIM + i) * W_DIM) + j];
        ssign[j * 256 + c] = (char)((val > 0.f) - (val < 0.f));
    }
    __syncthreads();
    for (int idx = t; idx < 8192; idx += 256) {
        int jj = idx >> 8, cc = idx & 255;
        int l = jj        ? ssign[(jj - 1) * 256 + cc] : 0;
        int m =             ssign[ jj      * 256 + cc];
        int r = (jj < 31) ? ssign[(jj + 1) * 256 + cc] : 0;
        sprod[idx] = (char)(l * m * r);
    }
    __syncthreads();
    size_t base = (size_t)blk * 8192;
    for (int k = t * 16; k < 8192; k += 256 * 16) {
        *(int4*)(xbp + base + k) = *(const int4*)&ssign[k];
        *(int4*)(xpp + base + k) = *(const int4*)&sprod[k];
    }
}

__global__ __launch_bounds__(256)
void maj3_gemm(const char* __restrict__ wp1, const char* __restrict__ wp2,
               const char* __restrict__ xbp, const char* __restrict__ xpp,
               float* __restrict__ out) {
    __shared__ char sm_xb[4 * 34 * 256];
    __shared__ char sm_xp[4 * 32 * 256];

    int blk = blockIdx.x;
    int ip  = blk & 15;
    int n   = blk >> 4;
    int rowBase = ip * 2;
    int tid  = threadIdx.x;
    int lane = tid & 31;
    int wv   = tid >> 5;

    const int4 z4 = {0, 0, 0, 0};

    for (int r = 0; r < 4; ++r) {
        int gr = rowBase + r - 1;
        char* xrow = &sm_xb[r * 34 * 256];
        char* prow = &sm_xp[r * 32 * 256];
        if (gr < 0 || gr >= H_DIM) {
            for (int k = tid; k < 544; k += 256) *(int4*)(xrow + k * 16) = z4;
            for (int k = tid; k < 512; k += 256) *(int4*)(prow + k * 16) = z4;
        } else {
            if (tid < 32) {
                int col = (tid >> 4) ? 33 : 0;
                *(int4*)(xrow + col * 256 + (tid & 15) * 16) = z4;
            }
            size_t gbase = ((size_t)(n * H_DIM + gr) * W_DIM) * 256;
            for (int k = tid; k < 512; k += 256) {
                int col = k >> 4, off = (k & 15) * 16;
                uint32_t lx = (uint32_t)(uintptr_t)(xrow + (col + 1) * 256 + off);
                uint64_t gx = (uint64_t)(uintptr_t)(xbp + gbase + col * 256 + off);
                asm volatile("global_load_async_to_lds_b128 %0, %1, off"
                             :: "v"(lx), "v"(gx) : "memory");
                uint32_t lp = (uint32_t)(uintptr_t)(prow + col * 256 + off);
                uint64_t gp = (uint64_t)(uintptr_t)(xpp + gbase + col * 256 + off);
                asm volatile("global_load_async_to_lds_b128 %0, %1, off"
                             :: "v"(lp), "v"(gp) : "memory");
            }
        }
    }
    asm volatile("s_wait_asynccnt 0" ::: "memory");
    __syncthreads();

    v8i zero = {0, 0, 0, 0, 0, 0, 0, 0};
    v8i acc0[4] = {zero, zero, zero, zero};
    v8i acc1[4] = {zero, zero, zero, zero};

    int wvU = __builtin_amdgcn_readfirstlane(wv);
    int ot0 = wvU * 2;
    int ot1 = wvU * 2 + 1;
    uint32_t laneOff = (uint32_t)lane * 32u;

    int nl = lane & 15;
    int lh = lane >> 4;
    int c0base = lh << 4;

    #pragma unroll 1
    for (int t = 0; t < 9; ++t) {
        int di = t / 3, dj = t % 3;
        const char* a0base = wp1 + ((size_t)(ot0 * 9 + t) * 4) * 1024;
        const char* a1base = wp1 + ((size_t)(ot1 * 9 + t) * 4) * 1024;
        #pragma unroll 1
        for (int q = 0; q < 4; ++q) {
            const char* a0p = a0base + (size_t)q * 1024;
            const char* a1p = a1base + (size_t)q * 1024;
            int4 a0l = *(const int4*)(a0p + laneOff);
            int4 a0h = *(const int4*)(a0p + laneOff + 16);
            int4 a1l = *(const int4*)(a1p + laneOff);
            int4 a1h = *(const int4*)(a1p + laneOff + 16);
            v8i A0 = {a0l.x, a0l.y, a0l.z, a0l.w, a0h.x, a0h.y, a0h.z, a0h.w};
            v8i A1 = {a1l.x, a1l.y, a1l.z, a1l.w, a1h.x, a1h.y, a1h.z, a1h.w};
            int c0 = q * 64 + c0base;
            #pragma unroll
            for (int s = 0; s < 4; ++s) {
                int row = (s >> 1) + di;
                int col = (s & 1) * 16 + nl + dj;
                const char* bp = &sm_xb[(row * 34 + col) * 256 + c0];
                int4 blo = *(const int4*)bp;
                int4 bhi = *(const int4*)(bp + 32);
                v8i B = {blo.x, blo.y, blo.z, blo.w, bhi.x, bhi.y, bhi.z, bhi.w};
                acc0[s] = __builtin_amdgcn_wmma_i32_16x16x64_iu8(
                    true, A0, true, B, acc0[s], false, false);
                acc1[s] = __builtin_amdgcn_wmma_i32_16x16x64_iu8(
                    true, A1, true, B, acc1[s], false, false);
            }
        }
    }

    #pragma unroll 1
    for (int r = 0; r < 3; ++r) {
        const char* a0base = wp2 + ((size_t)(ot0 * 3 + r) * 4) * 1024;
        const char* a1base = wp2 + ((size_t)(ot1 * 3 + r) * 4) * 1024;
        #pragma unroll 1
        for (int q = 0; q < 4; ++q) {
            const char* a0p = a0base + (size_t)q * 1024;
            const char* a1p = a1base + (size_t)q * 1024;
            int4 a0l = *(const int4*)(a0p + laneOff);
            int4 a0h = *(const int4*)(a0p + laneOff + 16);
            int4 a1l = *(const int4*)(a1p + laneOff);
            int4 a1h = *(const int4*)(a1p + laneOff + 16);
            v8i A0 = {a0l.x, a0l.y, a0l.z, a0l.w, a0h.x, a0h.y, a0h.z, a0h.w};
            v8i A1 = {a1l.x, a1l.y, a1l.z, a1l.w, a1h.x, a1h.y, a1h.z, a1h.w};
            int c0 = q * 64 + c0base;
            #pragma unroll
            for (int s = 0; s < 4; ++s) {
                int row = (s >> 1) + r;
                int col = (s & 1) * 16 + nl;
                const char* bp = &sm_xp[(row * 32 + col) * 256 + c0];
                int4 blo = *(const int4*)bp;
                int4 bhi = *(const int4*)(bp + 32);
                v8i B = {blo.x, blo.y, blo.z, blo.w, bhi.x, bhi.y, bhi.z, bhi.w};
                acc0[s] = __builtin_amdgcn_wmma_i32_16x16x64_iu8(
                    true, A0, true, B, acc0[s], false, false);
                acc1[s] = __builtin_amdgcn_wmma_i32_16x16x64_iu8(
                    true, A1, true, B, acc1[s], false, false);
            }
        }
    }

    #pragma unroll
    for (int s = 0; s < 4; ++s) {
        int i = rowBase + (s >> 1);
        int j = (s & 1) * 16 + nl;
        int ocB0 = ot0 * 16 + lh * 8;
        int ocB1 = ot1 * 16 + lh * 8;
        #pragma unroll
        for (int v = 0; v < 8; ++v) {
            out[(((size_t)(n * C_OUT + ocB0 + v) * H_DIM + i) * W_DIM) + j] =
                0.5f * (float)acc0[s][v];
            out[(((size_t)(n * C_OUT + ocB1 + v) * H_DIM + i) * W_DIM) + j] =
                0.5f * (float)acc1[s][v];
        }
    }
}

extern "C" void kernel_launch(void* const* d_in, const int* in_sizes, int n_in,
                              void* d_out, int out_size, void* d_ws, size_t ws_size,
                              hipStream_t stream) {
    const float* x = (const float*)d_in[0];
    const float* w = (const float*)d_in[1];
    float* out = (float*)d_out;

    char* ws  = (char*)d_ws;
    char* wp1 = ws;
    char* wp2 = wp1 + W1_BYTES;
    char* xbp = wp2 + W2_BYTES;
    char* xpp = xbp + XB_BYTES;

    maj3_pack_weights<<<768, 256, 0, stream>>>(w, wp1, wp2);
    maj3_pack_acts<<<N_IMG * H_DIM, 256, 0, stream>>>(x, xbp, xpp);
    maj3_gemm<<<N_IMG * 16, 256, 0, stream>>>(wp1, wp2, xbp, xpp, out);
}
